// EncoderBlock_51436528337579
// MI455X (gfx1250) — matrix operand order repaired and checked by reading, not compiled, not run on hardware
//
#include <hip/hip_runtime.h>
#include <stdint.h>

// ---------------------------------------------------------------------------
// Transformer encoder block (pre-norm) for MI455X / gfx1250, wave32 WMMA.
// All GEMMs + attention run on v_wmma_f32_16x16x32_bf16 (bf16 in, f32 acc).
// GEMM stages tiles with double-buffered async global->LDS copies (ASYNCcnt).
// ---------------------------------------------------------------------------

#define B_   4
#define S_   2048
#define DM_  1024
#define H_   16
#define DK_  64
#define DFF_ 4096
#define MTOT (B_ * S_)   // 8192 rows

#define AS1 __attribute__((address_space(1)))
#define AS3 __attribute__((address_space(3)))

#if defined(__has_builtin)
#if __has_builtin(__builtin_amdgcn_global_load_async_to_lds_b128) && \
    __has_builtin(__builtin_amdgcn_s_wait_asynccnt)
#define HAVE_ASYNC_LDS 1
#endif
#endif
#ifndef HAVE_ASYNC_LDS
#define HAVE_ASYNC_LDS 0
#endif

typedef __attribute__((ext_vector_type(16))) __bf16 bf16x16;
typedef __attribute__((ext_vector_type(8)))  float  floatx8;
typedef int i32x4 __attribute__((vector_size(16)));   // matches builtin pointee

union AFrag { bf16x16 v; uint32_t u[8]; };

static __device__ __forceinline__ uint16_t f2bf(float f) {
    uint32_t u = __float_as_uint(f);
    uint32_t r = (u + 0x7fffu + ((u >> 16) & 1u)) >> 16;   // RNE
    return (uint16_t)r;
}
// multiply both packed bf16 halves by 0.125 (exact: power of two)
static __device__ __forceinline__ uint32_t scale8th(uint32_t p) {
    float lo = __uint_as_float((p & 0xffffu) << 16) * 0.125f;
    float hi = __uint_as_float(p & 0xffff0000u) * 0.125f;
    return ((uint32_t)f2bf(hi) << 16) | (uint32_t)f2bf(lo);
}
// 16-byte LDS/global load into 4 dwords
static __device__ __forceinline__ void ld16(uint32_t* d, const uint16_t* p) {
    uint4 t = *reinterpret_cast<const uint4*>(p);
    d[0] = t.x; d[1] = t.y; d[2] = t.z; d[3] = t.w;
}
// 16-byte global -> LDS copy: async on gfx1250 toolchains that expose it
static __device__ __forceinline__ void cp16_g2l(const uint16_t* g, uint16_t* l) {
#if HAVE_ASYNC_LDS
    __builtin_amdgcn_global_load_async_to_lds_b128(
        (AS1 i32x4*)g, (AS3 i32x4*)l, 0, 0);
#else
    *reinterpret_cast<uint4*>(l) = *reinterpret_cast<const uint4*>(g);
#endif
}
static __device__ __forceinline__ void wait_async_le8() {
#if HAVE_ASYNC_LDS
    __builtin_amdgcn_s_wait_asynccnt(8);
#endif
}
static __device__ __forceinline__ void wait_async_0() {
#if HAVE_ASYNC_LDS
    __builtin_amdgcn_s_wait_asynccnt(0);
#endif
}

// ---------------------------------------------------------------------------
// f32 -> bf16 conversion (grid-stride)
// ---------------------------------------------------------------------------
__global__ void f32_to_bf16_kernel(const float* __restrict__ in,
                                   uint16_t* __restrict__ out, int n) {
    for (int i = blockIdx.x * blockDim.x + threadIdx.x; i < n;
         i += gridDim.x * blockDim.x)
        out[i] = f2bf(in[i]);
}

// ---------------------------------------------------------------------------
// LayerNorm over last dim (1024), ddof=1, (x-mean)/(std+eps); bf16 output.
// ---------------------------------------------------------------------------
__global__ __launch_bounds__(256)
void layernorm_bf16_kernel(const float* __restrict__ x,
                           const float* __restrict__ alpha,
                           const float* __restrict__ beta,
                           uint16_t* __restrict__ out) {
    const int row = blockIdx.x;
    const int t = threadIdx.x;
    const float* xr = x + (size_t)row * DM_;
    float4 v = reinterpret_cast<const float4*>(xr)[t];

    __shared__ float red[8];
    float s = v.x + v.y + v.z + v.w;
    #pragma unroll
    for (int off = 16; off; off >>= 1) s += __shfl_xor(s, off, 32);
    if ((t & 31) == 0) red[t >> 5] = s;
    __syncthreads();
    float tot = 0.f;
    #pragma unroll
    for (int i = 0; i < 8; i++) tot += red[i];
    const float mean = tot * (1.0f / 1024.0f);
    __syncthreads();

    float dx0 = v.x - mean, dx1 = v.y - mean, dx2 = v.z - mean, dx3 = v.w - mean;
    float sq = dx0 * dx0 + dx1 * dx1 + dx2 * dx2 + dx3 * dx3;
    #pragma unroll
    for (int off = 16; off; off >>= 1) sq += __shfl_xor(sq, off, 32);
    if ((t & 31) == 0) red[t >> 5] = sq;
    __syncthreads();
    float tot2 = 0.f;
    #pragma unroll
    for (int i = 0; i < 8; i++) tot2 += red[i];
    const float var = tot2 * (1.0f / 1023.0f);          // ddof = 1
    const float inv = 1.0f / (sqrtf(var) + 1e-6f);      // (std + eps) denom
    const float a = alpha[0], b = beta[0];

    uint16_t* orow = out + (size_t)row * DM_ + t * 4;
    orow[0] = f2bf(a * (dx0 * inv) + b);
    orow[1] = f2bf(a * (dx1 * inv) + b);
    orow[2] = f2bf(a * (dx2 * inv) + b);
    orow[3] = f2bf(a * (dx3 * inv) + b);
}

// ---------------------------------------------------------------------------
// Tiled bf16 GEMM: C[M,N] = A[M,K] @ B[K,N] + bias (+ epilogue)
//   EPI 0: out bf16 = acc + bias
//   EPI 1: out f32  = residual + acc + bias
//   EPI 2: out bf16 = relu(acc + bias)
// Block 128 threads (4 waves), block tile 128x128, wave tile 64x64, BK=32.
// Double-buffered LDS; async global->LDS staging (8 b128 ops/wave/tile).
// ---------------------------------------------------------------------------
template <int EPI>
__global__ __launch_bounds__(128)
void gemm_bf16_kernel(const uint16_t* __restrict__ A,
                      const uint16_t* __restrict__ Bm,
                      const float* __restrict__ bias,
                      const float* __restrict__ resid,
                      uint16_t* __restrict__ outBF,
                      float* __restrict__ outF,
                      int M, int N, int K) {
    __shared__ __align__(16) uint16_t sA[2][128 * 32];
    __shared__ __align__(16) uint16_t sB[2][32 * 128];

    const int tid   = threadIdx.x;
    const int wave  = tid >> 5, lane = tid & 31;
    const int waveM = wave >> 1, waveN = wave & 1;
    const int mlane = lane & 15, hf = lane >> 4;
    const int m0 = blockIdx.x * 128, n0 = blockIdx.y * 128;

    floatx8 acc[4][4];
    #pragma unroll
    for (int i = 0; i < 4; i++)
        #pragma unroll
        for (int j = 0; j < 4; j++)
            #pragma unroll
            for (int r = 0; r < 8; r++) acc[i][j][r] = 0.0f;

    // stage one 128x32 A slab + 32x128 B slab into buffer `buf`
    auto issue_tile = [&](int buf, int k0) {
        #pragma unroll
        for (int i = 0; i < 4; i++) {            // A tile 128x32
            int c = tid + 128 * i;
            int r = c >> 2, col = (c & 3) << 3;
            cp16_g2l(&A[(size_t)(m0 + r) * K + k0 + col],
                     &sA[buf][r * 32 + col]);
        }
        #pragma unroll
        for (int i = 0; i < 4; i++) {            // B tile 32x128
            int c = tid + 128 * i;
            int r = c >> 4, col = (c & 15) << 3;
            cp16_g2l(&Bm[(size_t)(k0 + r) * N + n0 + col],
                     &sB[buf][r * 128 + col]);
        }
    };

    const int nk = K >> 5;
    issue_tile(0, 0);
    for (int it = 0; it < nk; ++it) {
        const int cur = it & 1;
        if (it + 1 < nk) {
            issue_tile(cur ^ 1, (it + 1) << 5);  // prefetch next slab
            wait_async_le8();                    // current slab resident
        } else {
            wait_async_0();
        }
        __syncthreads();

        AFrag a[4], b[4];
        #pragma unroll
        for (int wm = 0; wm < 4; wm++) {
            // A frag: lane holds row m = lane%16; K halves per ISA layout
            int base = (waveM * 64 + wm * 16 + mlane) * 32 + hf * 8;
            ld16(a[wm].u, &sA[cur][base]);
            ld16(a[wm].u + 4, &sA[cur][base + 16]);
        }
        #pragma unroll
        for (int wn = 0; wn < 4; wn++) {
            // B frag: lane holds column N = lane%16; K halves per ISA layout
            int col = waveN * 64 + wn * 16 + mlane;
            #pragma unroll
            for (int i = 0; i < 8; i++) {
                int k = (i < 4) ? hf * 8 + 2 * i : 16 + hf * 8 + 2 * (i - 4);
                b[wn].u[i] = (uint32_t)sB[cur][k * 128 + col] |
                             ((uint32_t)sB[cur][(k + 1) * 128 + col] << 16);
            }
        }
        #pragma unroll
        for (int wm = 0; wm < 4; wm++)
            #pragma unroll
            for (int wn = 0; wn < 4; wn++)
                acc[wm][wn] = __builtin_amdgcn_wmma_f32_16x16x32_bf16(
                    false, a[wm].v, false, b[wn].v, (short)0, acc[wm][wn],
                    false, false);
        __syncthreads();   // protect buffer reuse by next iteration's issue
    }

    // epilogue: C layout M = r + 8*(lane/16), N = lane%16
    #pragma unroll
    for (int wn = 0; wn < 4; wn++) {
        const int n = n0 + waveN * 64 + wn * 16 + mlane;
        const float bv = bias[n];
        #pragma unroll
        for (int wm = 0; wm < 4; wm++) {
            #pragma unroll
            for (int r = 0; r < 8; r++) {
                const int m = m0 + waveM * 64 + wm * 16 + r + 8 * hf;
                const size_t idx = (size_t)m * N + n;
                float v = acc[wm][wn][r] + bv;
                if (EPI == 0) {
                    outBF[idx] = f2bf(v);
                } else if (EPI == 1) {
                    outF[idx] = resid[idx] + v;
                } else {
                    outBF[idx] = f2bf(v > 0.0f ? v : 0.0f);
                }
            }
        }
    }
}

// ---------------------------------------------------------------------------
// Flash attention: per block 4 waves x 16 queries (64 queries), one (b,head).
// Q pre-scaled by 1/8 (exact in bf16). Online softmax over key chunks of 32.
// Q/K/V/ctx stored row-major [B*S, 1024], column = head*64 + d.
// ---------------------------------------------------------------------------
__global__ __launch_bounds__(128)
void flash_attn_kernel(const uint16_t* __restrict__ Q,
                       const uint16_t* __restrict__ Kt,
                       const uint16_t* __restrict__ V,
                       const int* __restrict__ mask,
                       uint16_t* __restrict__ ctx) {
    __shared__ __align__(16) uint16_t sKT[64 * 32];      // [key][d]
    __shared__ __align__(16) uint16_t sV[32 * 64];       // [key][d]
    __shared__ __align__(16) uint16_t sP[4][16 * 32];    // per-wave probs
    __shared__ int sMask[32];

    const int tid  = threadIdx.x;
    const int wave = tid >> 5, lane = tid & 31;
    const int mlane = lane & 15, hf = lane >> 4;
    const int bh = blockIdx.y;
    const int b = bh >> 4, hh = bh & 15;
    const int q0 = blockIdx.x * 64 + wave * 16;

    // load + prescale Q fragments (16x64 -> two 16x32 A-frags)
    AFrag aq[2];
    const uint16_t* qrow =
        Q + ((size_t)(b * S_ + q0 + mlane)) * DM_ + hh * DK_;
    #pragma unroll
    for (int j = 0; j < 2; j++) {
        ld16(aq[j].u,     qrow + j * 32 + hf * 8);
        ld16(aq[j].u + 4, qrow + j * 32 + 16 + hf * 8);
        #pragma unroll
        for (int i = 0; i < 8; i++) aq[j].u[i] = scale8th(aq[j].u[i]);
    }

    float mrow[8], lsum[8];
    floatx8 cacc[4];
    #pragma unroll
    for (int r = 0; r < 8; r++) { mrow[r] = -1e30f; lsum[r] = 0.0f; }
    #pragma unroll
    for (int nt = 0; nt < 4; nt++)
        #pragma unroll
        for (int r = 0; r < 8; r++) cacc[nt][r] = 0.0f;

    const int key = tid & 31, dbase = (tid >> 5) * 16;
    for (int kc = 0; kc < S_ / 32; kc++) {
        __syncthreads();
        {   // stage K chunk row-major: sKT[key][d]
            const uint16_t* krow =
                Kt + ((size_t)(b * S_ + kc * 32 + key)) * DM_ + hh * DK_ + dbase;
            uint32_t tmp[8];
            ld16(tmp, krow);
            ld16(tmp + 4, krow + 8);
            #pragma unroll
            for (int e = 0; e < 16; e++) {
                uint16_t val = (e & 1) ? (uint16_t)(tmp[e >> 1] >> 16)
                                       : (uint16_t)(tmp[e >> 1] & 0xffffu);
                sKT[key * 64 + dbase + e] = val;
            }
        }
        {   // stage V chunk row-major: sV[key][d]
            const uint16_t* vrow =
                V + ((size_t)(b * S_ + kc * 32 + key)) * DM_ + hh * DK_ + dbase;
            *reinterpret_cast<uint4*>(&sV[key * 64 + dbase]) =
                *reinterpret_cast<const uint4*>(vrow);
            *reinterpret_cast<uint4*>(&sV[key * 64 + dbase + 8]) =
                *reinterpret_cast<const uint4*>(vrow + 8);
        }
        if (tid < 32) sMask[tid] = mask[b * S_ + kc * 32 + tid];
        if (kc + 1 < S_ / 32) {  // prefetch next chunk (global_prefetch_b8)
            const size_t nrow = (size_t)(b * S_ + (kc + 1) * 32 + key) * DM_ +
                                hh * DK_ + dbase;
            __builtin_prefetch(Kt + nrow, 0, 0);
            __builtin_prefetch(V + nrow, 0, 0);
        }
        __syncthreads();

        // scores (already /8 via Q prescale): two 16x16 tiles of keys
        floatx8 sc[2];
        #pragma unroll
        for (int kt = 0; kt < 2; kt++)
            #pragma unroll
            for (int r = 0; r < 8; r++) sc[kt][r] = 0.0f;
        #pragma unroll
        for (int kt = 0; kt < 2; kt++) {
            #pragma unroll
            for (int j = 0; j < 2; j++) {
                AFrag bk;
                int base = (kt * 16 + mlane) * 64 + j * 32 + hf * 8;
                ld16(bk.u, &sKT[base]);
                ld16(bk.u + 4, &sKT[base + 16]);
                sc[kt] = __builtin_amdgcn_wmma_f32_16x16x32_bf16(
                    false, aq[j].v, false, bk.v, (short)0, sc[kt], false, false);
            }
            const int mv = sMask[kt * 16 + mlane];
            if (mv == 0) {
                #pragma unroll
                for (int r = 0; r < 8; r++) sc[kt][r] = -1e9f;
            }
        }

        // online softmax (row M = r + 8*hf lives across lanes of one half)
        float cm[8];
        #pragma unroll
        for (int r = 0; r < 8; r++) cm[r] = fmaxf(sc[0][r], sc[1][r]);
        #pragma unroll
        for (int off = 8; off; off >>= 1)
            #pragma unroll
            for (int r = 0; r < 8; r++)
                cm[r] = fmaxf(cm[r], __shfl_xor(cm[r], off, 32));
        #pragma unroll
        for (int r = 0; r < 8; r++) {
            const float mnew = fmaxf(mrow[r], cm[r]);
            const float scl = __expf(mrow[r] - mnew);
            mrow[r] = mnew;
            lsum[r] *= scl;
            #pragma unroll
            for (int nt = 0; nt < 4; nt++) cacc[nt][r] *= scl;
        }

        float psum[8];
        #pragma unroll
        for (int r = 0; r < 8; r++) psum[r] = 0.0f;
        #pragma unroll
        for (int kt = 0; kt < 2; kt++)
            #pragma unroll
            for (int r = 0; r < 8; r++) {
                const float p = __expf(sc[kt][r] - mrow[r]);
                psum[r] += p;
                sP[wave][(r + 8 * hf) * 32 + kt * 16 + mlane] = f2bf(p);
            }
        #pragma unroll
        for (int off = 8; off; off >>= 1)
            #pragma unroll
            for (int r = 0; r < 8; r++)
                psum[r] += __shfl_xor(psum[r], off, 32);
        #pragma unroll
        for (int r = 0; r < 8; r++) lsum[r] += psum[r];

        __syncthreads();   // sP write -> A-frag read re-layout

        // ctx += P(16x32) @ V(32x64)
        AFrag ap;
        {
            int base = mlane * 32 + hf * 8;
            ld16(ap.u, &sP[wave][base]);
            ld16(ap.u + 4, &sP[wave][base + 16]);
        }
        #pragma unroll
        for (int nt = 0; nt < 4; nt++) {
            AFrag bv;
            int col = nt * 16 + mlane;
            #pragma unroll
            for (int i = 0; i < 8; i++) {
                int k = (i < 4) ? hf * 8 + 2 * i : 16 + hf * 8 + 2 * (i - 4);
                bv.u[i] = (uint32_t)sV[k * 64 + col] |
                          ((uint32_t)sV[(k + 1) * 64 + col] << 16);
            }
            cacc[nt] = __builtin_amdgcn_wmma_f32_16x16x32_bf16(
                false, ap.v, false, bv.v, (short)0, cacc[nt], false, false);
        }
    }

    // finalize: divide by row sums, write ctx (bf16)
    float inv[8];
    #pragma unroll
    for (int r = 0; r < 8; r++) inv[r] = 1.0f / lsum[r];
    #pragma unroll
    for (int nt = 0; nt < 4; nt++) {
        #pragma unroll
        for (int r = 0; r < 8; r++) {
            const int row = b * S_ + q0 + r + 8 * hf;
            const int col = hh * DK_ + nt * 16 + mlane;
            ctx[(size_t)row * DM_ + col] = f2bf(cacc[nt][r] * inv[r]);
        }
    }
}

// ---------------------------------------------------------------------------
// Host-side orchestration
// ---------------------------------------------------------------------------
extern "C" void kernel_launch(void* const* d_in, const int* in_sizes, int n_in,
                              void* d_out, int out_size, void* d_ws,
                              size_t ws_size, hipStream_t stream) {
    (void)in_sizes; (void)n_in; (void)out_size; (void)ws_size;
    const float* x    = (const float*)d_in[0];
    const int*   mask = (const int*)d_in[1];
    const float* Wq = (const float*)d_in[2];  const float* bq = (const float*)d_in[3];
    const float* Wk = (const float*)d_in[4];  const float* bk = (const float*)d_in[5];
    const float* Wv = (const float*)d_in[6];  const float* bv = (const float*)d_in[7];
    const float* Wo = (const float*)d_in[8];  const float* bo = (const float*)d_in[9];
    const float* W1 = (const float*)d_in[10]; const float* b1 = (const float*)d_in[11];
    const float* W2 = (const float*)d_in[12]; const float* b2 = (const float*)d_in[13];
    const float* ln1a = (const float*)d_in[14]; const float* ln1b = (const float*)d_in[15];
    const float* ln2a = (const float*)d_in[16]; const float* ln2b = (const float*)d_in[17];
    float* out = (float*)d_out;

    const size_t SZ_W   = (size_t)DM_ * DM_;    // 1M elems
    const size_t SZ_WF  = (size_t)DM_ * DFF_;   // 4M elems
    const size_t SZ_ACT = (size_t)MTOT * DM_;   // 8M elems
    const size_t SZ_FF  = (size_t)MTOT * DFF_;  // 32M elems

    char* p = (char*)d_ws;
    uint16_t* wq_bf  = (uint16_t*)p; p += SZ_W * 2;
    uint16_t* wk_bf  = (uint16_t*)p; p += SZ_W * 2;
    uint16_t* wv_bf  = (uint16_t*)p; p += SZ_W * 2;
    uint16_t* wo_bf  = (uint16_t*)p; p += SZ_W * 2;
    uint16_t* w1_bf  = (uint16_t*)p; p += SZ_WF * 2;
    uint16_t* w2_bf  = (uint16_t*)p; p += SZ_WF * 2;
    uint16_t* h_bf   = (uint16_t*)p; p += SZ_ACT * 2;
    uint16_t* q_bf   = (uint16_t*)p; p += SZ_ACT * 2;
    uint16_t* k_bf   = (uint16_t*)p; p += SZ_ACT * 2;
    uint16_t* v_bf   = (uint16_t*)p; p += SZ_ACT * 2;
    uint16_t* ctx_bf = (uint16_t*)p; p += SZ_ACT * 2;
    float*    x1     = (float*)p;    p += SZ_ACT * 4;
    uint16_t* h2_bf  = (uint16_t*)p; p += SZ_ACT * 2;
    uint16_t* ff1_bf = (uint16_t*)p; p += SZ_FF * 2;

    // 0) weights -> bf16
    f32_to_bf16_kernel<<<2048, 256, 0, stream>>>(Wq, wq_bf, (int)SZ_W);
    f32_to_bf16_kernel<<<2048, 256, 0, stream>>>(Wk, wk_bf, (int)SZ_W);
    f32_to_bf16_kernel<<<2048, 256, 0, stream>>>(Wv, wv_bf, (int)SZ_W);
    f32_to_bf16_kernel<<<2048, 256, 0, stream>>>(Wo, wo_bf, (int)SZ_W);
    f32_to_bf16_kernel<<<4096, 256, 0, stream>>>(W1, w1_bf, (int)SZ_WF);
    f32_to_bf16_kernel<<<4096, 256, 0, stream>>>(W2, w2_bf, (int)SZ_WF);

    // 1) h = LN1(x)
    layernorm_bf16_kernel<<<MTOT, 256, 0, stream>>>(x, ln1a, ln1b, h_bf);

    // 2) Q/K/V projections
    dim3 gP(MTOT / 128, DM_ / 128);
    gemm_bf16_kernel<0><<<gP, 128, 0, stream>>>(h_bf, wq_bf, bq, nullptr,
                                                q_bf, nullptr, MTOT, DM_, DM_);
    gemm_bf16_kernel<0><<<gP, 128, 0, stream>>>(h_bf, wk_bf, bk, nullptr,
                                                k_bf, nullptr, MTOT, DM_, DM_);
    gemm_bf16_kernel<0><<<gP, 128, 0, stream>>>(h_bf, wv_bf, bv, nullptr,
                                                v_bf, nullptr, MTOT, DM_, DM_);

    // 3) attention
    dim3 gA(S_ / 64, B_ * H_);
    flash_attn_kernel<<<gA, 128, 0, stream>>>(q_bf, k_bf, v_bf, mask, ctx_bf);

    // 4) x1 = x + ctx @ Wo + bo
    gemm_bf16_kernel<1><<<gP, 128, 0, stream>>>(ctx_bf, wo_bf, bo, x,
                                                nullptr, x1, MTOT, DM_, DM_);

    // 5) h2 = LN2(x1)
    layernorm_bf16_kernel<<<MTOT, 256, 0, stream>>>(x1, ln2a, ln2b, h2_bf);

    // 6) ff1 = relu(h2 @ W1 + b1)
    dim3 gF1(MTOT / 128, DFF_ / 128);
    gemm_bf16_kernel<2><<<gF1, 128, 0, stream>>>(h2_bf, w1_bf, b1, nullptr,
                                                 ff1_bf, nullptr, MTOT, DFF_, DM_);

    // 7) out = x1 + ff1 @ W2 + b2
    gemm_bf16_kernel<1><<<gP, 128, 0, stream>>>(ff1_bf, w2_bf, b2, x1,
                                                nullptr, out, MTOT, DM_, DFF_);
}
